// TriEDNet_65077344469204
// MI455X (gfx1250) — matrix operand order repaired and checked by reading, not compiled, not run on hardware
//
#include <hip/hip_runtime.h>
#include <cmath>

typedef __bf16 bf16_t;
typedef __attribute__((ext_vector_type(16))) __bf16 v16bf;
typedef __attribute__((ext_vector_type(8)))  __bf16 v8bf;
typedef __attribute__((ext_vector_type(8)))  float  v8f;
typedef int v4i_vs __attribute__((vector_size(16)));

#define BM 128
#define BN 128
#define BK 32
#define LDT 40   // padded LDS tile stride (elements); 80 B = multiple of 16 B

#if defined(__HIP_DEVICE_COMPILE__) && defined(__has_builtin)
#if __has_builtin(__builtin_amdgcn_global_load_async_to_lds_b128)
#define USE_ASYNC 1
#endif
#endif
#ifndef USE_ASYNC
#define USE_ASYNC 0
#endif

#if USE_ASYNC
__device__ __forceinline__ void wait_async0() {
#if __has_builtin(__builtin_amdgcn_s_wait_asynccnt)
    __builtin_amdgcn_s_wait_asynccnt(0);
#else
    asm volatile("s_wait_asynccnt 0" ::: "memory");
#endif
}
#endif

// ---------------------------------------------------------------------------
// Tiled bf16 WMMA GEMM:  C = epilogue( A[MxK] @ W[KxN] ),  W given TRANSPOSED
// as Wt[N x K] (row-major, ldw == K).  Double-buffered LDS; async loads when
// available.  epilogue: v = acc; v += bias[col]; v = act(v); v *= alpha;
// v -= sub[r,c]; if(accumulate) v += Cf_old; store f32 (Cf) and/or bf16 (Cb).
// ---------------------------------------------------------------------------
__global__ __launch_bounds__(256) void gemm_bf16_kernel(
    const bf16_t* __restrict__ A, int lda,
    const bf16_t* __restrict__ Wt, int ldw,
    const float* __restrict__ bias,
    const float* __restrict__ sub, int ldsub,
    float* __restrict__ Cf, int ldcf,
    bf16_t* __restrict__ Cb, int ldcb,
    int M, int N, int K,
    float alpha, int act, int accumulate)
{
    __shared__ bf16_t As[2][BM * LDT];   // 128x32, row-major (K minor)
    __shared__ bf16_t Bs[2][BN * LDT];   // 128x32, n-major   (K minor)

    const int tid  = threadIdx.x;
    const int lane = tid & 31;
    const int wave = tid >> 5;
    const int wr   = wave >> 2;   // 0..1  -> 64-row half
    const int wc   = wave & 3;    // 0..3  -> 32-col quarter
    const int lm   = lane & 15;
    const int lh   = lane >> 4;
    const int m0   = blockIdx.y * BM;
    const int n0   = blockIdx.x * BN;
    (void)M;

    v8f acc[4][2];
#pragma unroll
    for (int i = 0; i < 4; ++i)
#pragma unroll
        for (int j = 0; j < 2; ++j)
#pragma unroll
            for (int q = 0; q < 8; ++q) acc[i][j][q] = 0.0f;

    // ---- tile loader: 128 rows x 32 cols each for A and Wt, 16B per op
    auto load_tile = [&](int k0, int s) {
#pragma unroll
        for (int p = 0; p < 2; ++p) {
            int idx = tid + p * 256;
            int r   = idx >> 2;            // 0..127
            int c8  = (idx & 3) * 8;       // 0,8,16,24
            const bf16_t* ga = A + (size_t)(m0 + r) * lda + (k0 + c8);
            bf16_t*       la = &As[s][r * LDT + c8];
            int br = n0 + r; if (br > N - 1) br = N - 1;   // clamp (cols>=N discarded)
            const bf16_t* gb = Wt + (size_t)br * ldw + (k0 + c8);
            bf16_t*       lb = &Bs[s][r * LDT + c8];
#if USE_ASYNC
            __builtin_amdgcn_global_load_async_to_lds_b128((v4i_vs*)ga, (v4i_vs*)la, 0, 0);
            __builtin_amdgcn_global_load_async_to_lds_b128((v4i_vs*)gb, (v4i_vs*)lb, 0, 0);
#else
            *(v8bf*)la = *(const v8bf*)ga;
            *(v8bf*)lb = *(const v8bf*)gb;
#endif
        }
    };

    load_tile(0, 0);
#if USE_ASYNC
    wait_async0();
#endif
    __syncthreads();

    int s = 0;
    for (int k0 = 0; k0 < K; k0 += BK, s ^= 1) {
        bool more = (k0 + BK) < K;
        if (more) load_tile(k0 + BK, s ^ 1);   // prefetch overlaps WMMA below

        // ---- fragments (ISA 16-bit layouts)
        v16bf afr[4], bfr[2];
#pragma unroll
        for (int i = 0; i < 4; ++i) {
            const bf16_t* pa = &As[s][(wr * 64 + i * 16 + lm) * LDT];
            v8bf lo = *(const v8bf*)&pa[lh * 8];        // K = lh*8 .. +7
            v8bf hi = *(const v8bf*)&pa[16 + lh * 8];   // K = 16+lh*8 .. +7
#pragma unroll
            for (int q = 0; q < 8; ++q) { afr[i][q] = lo[q]; afr[i][q + 8] = hi[q]; }
        }
#pragma unroll
        for (int j = 0; j < 2; ++j) {
            const bf16_t* pb = &Bs[s][(wc * 32 + j * 16 + lm) * LDT];
            v8bf lo = *(const v8bf*)&pb[lh * 8];        // K = lh*8 .. +7
            v8bf hi = *(const v8bf*)&pb[16 + lh * 8];   // K = 16+lh*8 .. +7
#pragma unroll
            for (int q = 0; q < 8; ++q) { bfr[j][q] = lo[q]; bfr[j][q + 8] = hi[q]; }
        }

#pragma unroll
        for (int i = 0; i < 4; ++i)
#pragma unroll
            for (int j = 0; j < 2; ++j)
                acc[i][j] = __builtin_amdgcn_wmma_f32_16x16x32_bf16(
                    false, afr[i], false, bfr[j], (short)0, acc[i][j], false, false);

        if (more) {
#if USE_ASYNC
            wait_async0();
#endif
            __syncthreads();
        }
    }

    // ---- epilogue
#pragma unroll
    for (int i = 0; i < 4; ++i) {
#pragma unroll
        for (int j = 0; j < 2; ++j) {
            int col = n0 + wc * 32 + j * 16 + lm;
            if (col >= N) continue;
#pragma unroll
            for (int r = 0; r < 8; ++r) {
                int row = m0 + wr * 64 + i * 16 + lh * 8 + r;
                float v = acc[i][j][r];
                if (bias) v += bias[col];
                if (act == 1) v = fmaxf(v, 0.0f);
                else if (act == 2) v = tanhf(v);
                v *= alpha;
                if (sub) v -= sub[(size_t)row * ldsub + col];
                if (accumulate) v += Cf[(size_t)row * ldcf + col];
                if (Cf) Cf[(size_t)row * ldcf + col] = v;
                if (Cb) Cb[(size_t)row * ldcb + col] = (bf16_t)v;
            }
        }
    }
}

// ---------------------------------------------------------------------------
// small helper kernels
// ---------------------------------------------------------------------------
__global__ void f32_to_bf16_kernel(const float* __restrict__ src,
                                   bf16_t* __restrict__ dst, size_t n)
{
    size_t i  = (size_t)blockIdx.x * blockDim.x + threadIdx.x;
    size_t st = (size_t)gridDim.x * blockDim.x;
    for (; i < n; i += st) dst[i] = (bf16_t)src[i];
}

__global__ void f32_to_bf16_T_kernel(const float* __restrict__ src,
                                     bf16_t* __restrict__ dst, int R, int C)
{
    int i = blockIdx.x * blockDim.x + threadIdx.x;
    if (i < R * C) {
        int r = i / C, c = i % C;
        dst[(size_t)c * R + r] = (bf16_t)src[(size_t)r * C + c];
    }
}

__global__ void proto_norm_kernel(const float* __restrict__ proto,
                                  float* __restrict__ pn, int P, int D)
{
    int j = blockIdx.x * blockDim.x + threadIdx.x;
    if (j < P) {
        float s = 0.0f;
        for (int d = 0; d < D; ++d) { float v = proto[(size_t)j * D + d]; s += v * v; }
        pn[j] = s;
    }
}

// layer norm over rows of width W (256 or 512); block = 256 threads
__global__ __launch_bounds__(256) void layernorm_kernel(
    const float* __restrict__ in, const float* __restrict__ g,
    const float* __restrict__ b, int W,
    float* __restrict__ out_f,
    bf16_t* __restrict__ out_b1, int ld1,
    bf16_t* __restrict__ out_b2, int ld2)
{
    int row = blockIdx.x, tid = threadIdx.x;
    const float* x = in + (size_t)row * W;
    bool two = (W > 256);
    float x0 = x[tid];
    float x1 = two ? x[tid + 256] : 0.0f;
    __shared__ float red[256];
    red[tid] = x0 + x1; __syncthreads();
    for (int s = 128; s > 0; s >>= 1) { if (tid < s) red[tid] += red[tid + s]; __syncthreads(); }
    float mean = red[0] / (float)W; __syncthreads();
    float d0 = x0 - mean, d1 = two ? (x1 - mean) : 0.0f;
    red[tid] = d0 * d0 + d1 * d1; __syncthreads();
    for (int s = 128; s > 0; s >>= 1) { if (tid < s) red[tid] += red[tid + s]; __syncthreads(); }
    float rstd = rsqrtf(red[0] / (float)W + 1e-5f);
    {
        float y = d0 * rstd * g[tid] + b[tid];
        if (out_f)  out_f [(size_t)row * W   + tid] = y;
        if (out_b1) out_b1[(size_t)row * ld1 + tid] = (bf16_t)y;
        if (out_b2) out_b2[(size_t)row * ld2 + tid] = (bf16_t)y;
    }
    if (two) {
        int d = tid + 256;
        float y = d1 * rstd * g[d] + b[d];
        if (out_f)  out_f [(size_t)row * W   + d] = y;
        if (out_b1) out_b1[(size_t)row * ld1 + d] = (bf16_t)y;
        if (out_b2) out_b2[(size_t)row * ld2 + d] = (bf16_t)y;
    }
}

// softmax over rows of width 512 with 1/max(tau,0.1) scale; bf16 out
__global__ __launch_bounds__(256) void softmax_kernel(
    const float* __restrict__ in, const float* __restrict__ tau,
    bf16_t* __restrict__ out)
{
    const int W = 512;
    int row = blockIdx.x, tid = threadIdx.x;
    const float* x = in + (size_t)row * W;
    float it = 1.0f / fmaxf(tau[0], 0.1f);
    float a = x[tid] * it, b = x[tid + 256] * it;
    __shared__ float red[256];
    red[tid] = fmaxf(a, b); __syncthreads();
    for (int s = 128; s > 0; s >>= 1) { if (tid < s) red[tid] = fmaxf(red[tid], red[tid + s]); __syncthreads(); }
    float m = red[0]; __syncthreads();
    float ea = expf(a - m), eb = expf(b - m);
    red[tid] = ea + eb; __syncthreads();
    for (int s = 128; s > 0; s >>= 1) { if (tid < s) red[tid] += red[tid + s]; __syncthreads(); }
    float inv = 1.0f / red[0];
    out[(size_t)row * W + tid]       = (bf16_t)(ea * inv);
    out[(size_t)row * W + tid + 256] = (bf16_t)(eb * inv);
}

// effort = sigmoid( relu_hidden[row,:64] . w2 + b2 )
__global__ void effort_kernel(const float* __restrict__ gh,
                              const float* __restrict__ w2,
                              const float* __restrict__ b2,
                              float* __restrict__ effort, int Bsz, int H)
{
    int i = blockIdx.x * blockDim.x + threadIdx.x;
    if (i < Bsz) {
        float a = b2[0];
        for (int k = 0; k < H; ++k) a += gh[(size_t)i * H + k] * w2[k];
        float e = 1.0f / (1.0f + expf(-a));
        effort[i] = fminf(fmaxf(e, 0.0f), 1.0f);
    }
}

// argmin_j ( pn[j] - 2*dots[row,j] )  over P=512
__global__ __launch_bounds__(256) void argmin_kernel(
    const float* __restrict__ dots, const float* __restrict__ pn,
    int* __restrict__ jmin, int P)
{
    int row = blockIdx.x, tid = threadIdx.x;
    const float* d = dots + (size_t)row * P;
    __shared__ float sv[256]; __shared__ int si[256];
    float bv = 3.4e38f; int bi = 0;
    for (int j = tid; j < P; j += 256) {
        float v = pn[j] - 2.0f * d[j];
        if (v < bv) { bv = v; bi = j; }
    }
    sv[tid] = bv; si[tid] = bi; __syncthreads();
    for (int s = 128; s > 0; s >>= 1) {
        if (tid < s) {
            if (sv[tid + s] < sv[tid] ||
                (sv[tid + s] == sv[tid] && si[tid + s] < si[tid])) {
                sv[tid] = sv[tid + s]; si[tid] = si[tid + s];
            }
        }
        __syncthreads();
    }
    if (tid == 0) jmin[row] = si[0];
}

// h -= eta * ( gbuf + 0.4*(h - p*) + (2e-4/512)*h ); writes f32 + bf16 h
__global__ void update_kernel(float* __restrict__ h, bf16_t* __restrict__ hb,
                              const float* __restrict__ gbuf,
                              const int* __restrict__ jmin,
                              const float* __restrict__ proto,
                              const float* __restrict__ effort,
                              const float* __restrict__ step_sizes,
                              int t, float decay)
{
    size_t i = (size_t)blockIdx.x * blockDim.x + threadIdx.x;  // B*512
    int row = (int)(i >> 9);
    int d   = (int)(i & 511);
    float hv = h[i];
    float g  = gbuf[i];
    float p  = proto[(size_t)jmin[row] * 512 + d];
    g += 0.4f * (hv - p) + 3.90625e-7f * hv;          // 2*LAM_PROTO, 2*LAM_H/HID
    float eta = fminf(fmaxf(step_sizes[t], 1e-4f), 0.5f)
              * (0.5f + 0.5f * decay) * (0.5f + effort[row]);
    hv -= eta * g;
    h[i]  = hv;
    hb[i] = (bf16_t)hv;
}

// out = 0.6*lr + 0.4*( mn - log(sum_q exp(mn - d2_q)) ), d2 = ||h||^2+pn-2dots
__global__ __launch_bounds__(128) void final_kernel(
    const float* __restrict__ h, const float* __restrict__ dots,
    const float* __restrict__ pn, const float* __restrict__ lr,
    float* __restrict__ out)
{
    int row = blockIdx.x, tid = threadIdx.x;   // 128 threads
    const float* hr = h + (size_t)row * 512;
    __shared__ float red[128];
    float s = 0.0f;
#pragma unroll
    for (int q = 0; q < 4; ++q) { float v = hr[tid + q * 128]; s += v * v; }
    red[tid] = s; __syncthreads();
    for (int st = 64; st > 0; st >>= 1) { if (tid < st) red[tid] += red[tid + st]; __syncthreads(); }
    float hh = red[0];
    const float* dr = dots + (size_t)row * 512;
    int b4 = tid * 4;
    float d2[4];
#pragma unroll
    for (int q = 0; q < 4; ++q) d2[q] = hh + pn[b4 + q] - 2.0f * dr[b4 + q];
    float mn = fminf(fminf(d2[0], d2[1]), fminf(d2[2], d2[3]));
    float sum = 0.0f;
#pragma unroll
    for (int q = 0; q < 4; ++q) sum += expf(mn - d2[q]);
    float pl = mn - logf(sum);
    out[(size_t)row * 128 + tid] = 0.6f * lr[(size_t)row * 128 + tid] + 0.4f * pl;
}

// ---------------------------------------------------------------------------
extern "C" void kernel_launch(void* const* d_in, const int* in_sizes, int n_in,
                              void* d_out, int out_size, void* d_ws, size_t ws_size,
                              hipStream_t stream)
{
    (void)in_sizes; (void)n_in; (void)out_size; (void)ws_size;
    const int Bsz = 16384, DIN = 512, HID = 512, CODE = 256, CBSZ = 512;
    const int NCLS = 128, P = 512, GATEH = 64, FEAT = HID + NCLS; // 640

    const float* x        = (const float*)d_in[0];
    const float* enc_w1   = (const float*)d_in[1];
    const float* enc_b1   = (const float*)d_in[2];
    const float* enc_w2   = (const float*)d_in[3];
    const float* enc_b2   = (const float*)d_in[4];
    const float* enc_ln_g = (const float*)d_in[5];
    const float* enc_ln_b = (const float*)d_in[6];
    const float* sym_w    = (const float*)d_in[7];
    const float* sym_b    = (const float*)d_in[8];
    const float* sym_ln_g = (const float*)d_in[9];
    const float* sym_ln_b = (const float*)d_in[10];
    const float* cb_w     = (const float*)d_in[11];
    const float* cb_b     = (const float*)d_in[12];
    const float* codebook = (const float*)d_in[13];
    const float* tau      = (const float*)d_in[14];
    const float* align_w  = (const float*)d_in[15];
    const float* align_b  = (const float*)d_in[16];
    const float* drift_w1 = (const float*)d_in[17];
    const float* drift_b1 = (const float*)d_in[18];
    const float* drift_w2 = (const float*)d_in[19];
    const float* drift_b2 = (const float*)d_in[20];
    const float* steps    = (const float*)d_in[21];
    const float* proto    = (const float*)d_in[22];
    const float* head_w   = (const float*)d_in[23];
    const float* head_b   = (const float*)d_in[24];
    const float* gate_w1  = (const float*)d_in[25];
    const float* gate_b1  = (const float*)d_in[26];
    const float* gate_w2  = (const float*)d_in[27];
    const float* gate_b2  = (const float*)d_in[28];
    float* out = (float*)d_out;

    // ---- workspace layout (256B-aligned cursor)
    char* base = (char*)d_ws;
    size_t cur = 0;
    auto alloc = [&](size_t bytes) -> char* {
        cur = (cur + 255) & ~(size_t)255;
        char* p = base + cur; cur += bytes; return p;
    };
    // all weights stored TRANSPOSED [N x K] bf16 for direct n-major tile loads
    bf16_t* enc_w1T  = (bf16_t*)alloc((size_t)HID * DIN * 2);    // [HID x DIN]
    bf16_t* enc_w2T  = (bf16_t*)alloc((size_t)HID * HID * 2);
    bf16_t* sym_wT   = (bf16_t*)alloc((size_t)CODE * DIN * 2);   // [CODE x DIN]
    bf16_t* cb_wT    = (bf16_t*)alloc((size_t)CBSZ * CODE * 2);  // [CBSZ x CODE]
    bf16_t* cbookT   = (bf16_t*)alloc((size_t)CODE * CBSZ * 2);  // [CODE x CBSZ]
    bf16_t* align_fT = (bf16_t*)alloc((size_t)CODE * HID * 2);   // fwd: [CODE x HID]
    bf16_t* align_bT = (bf16_t*)alloc((size_t)HID * CODE * 2);   // bwd: [HID x CODE] = align_w
    bf16_t* drift1T  = (bf16_t*)alloc((size_t)HID * HID * 2);
    bf16_t* drift2T  = (bf16_t*)alloc((size_t)HID * HID * 2);
    bf16_t* head_wT  = (bf16_t*)alloc((size_t)NCLS * HID * 2);   // [NCLS x HID]
    bf16_t* gate_w1T = (bf16_t*)alloc((size_t)GATEH * FEAT * 2); // [GATEH x FEAT]
    bf16_t* protoB   = (bf16_t*)alloc((size_t)P * HID * 2);      // dots: [P x HID] = proto
    float*  pn       = (float*) alloc((size_t)P * 4);

    bf16_t* xb    = (bf16_t*)alloc((size_t)Bsz * DIN * 2);
    bf16_t* t0r   = (bf16_t*)alloc((size_t)Bsz * HID * 2);   // t0 bf16, later r bf16
    bf16_t* hb    = (bf16_t*)alloc((size_t)Bsz * HID * 2);
    bf16_t* featb = (bf16_t*)alloc((size_t)Bsz * FEAT * 2);
    bf16_t* hp_bf = (bf16_t*)alloc((size_t)Bsz * CODE * 2);
    bf16_t* piu   = (bf16_t*)alloc((size_t)Bsz * CBSZ * 2);  // pi bf16, later u bf16

    float* h_f   = (float*)alloc((size_t)Bsz * HID * 4);
    float* tcd   = (float*)alloc((size_t)Bsz * 512 * 4);     // t1 / cb-logits / dots
    float* gbuf  = (float*)alloc((size_t)Bsz * HID * 4);     // align-grad + drift
    float* s_f   = (float*)alloc((size_t)Bsz * CODE * 4);    // hs pre-LN, later s
    float* gh_f  = (float*)alloc((size_t)Bsz * GATEH * 4);
    float* lr_f  = (float*)alloc((size_t)Bsz * NCLS * 4);
    float* eff   = (float*)alloc((size_t)Bsz * 4);
    int*   jmin  = (int*)  alloc((size_t)Bsz * 4);

    auto conv = [&](const float* s, bf16_t* d, size_t n) {
        int blocks = (int)((n + 255) / 256); if (blocks > 4096) blocks = 4096;
        f32_to_bf16_kernel<<<blocks, 256, 0, stream>>>(s, d, n);
    };
    auto convT = [&](const float* s, bf16_t* d, int R, int C) {  // src[RxC] -> dst[CxR]
        f32_to_bf16_T_kernel<<<(R * C + 255) / 256, 256, 0, stream>>>(s, d, R, C);
    };
    auto gemm = [&](const bf16_t* A, int lda, const bf16_t* WtB,
                    const float* bias, const float* sub, int ldsub,
                    float* Cf, int ldcf, bf16_t* Cb, int ldcb,
                    int N, int K, float alpha, int act, int accf) {
        dim3 grid((N + BN - 1) / BN, Bsz / BM);
        gemm_bf16_kernel<<<grid, 256, 0, stream>>>(A, lda, WtB, K, bias,
            sub, ldsub, Cf, ldcf, Cb, ldcb, Bsz, N, K, alpha, act, accf);
    };

    // ---- weight / input conversions (deterministic, every call)
    convT(enc_w1,   enc_w1T,  DIN,  HID);   // [HID x DIN]
    convT(enc_w2,   enc_w2T,  HID,  HID);
    convT(sym_w,    sym_wT,   DIN,  CODE);  // [CODE x DIN]
    convT(cb_w,     cb_wT,    CODE, CBSZ);  // [CBSZ x CODE]
    convT(codebook, cbookT,   CBSZ, CODE);  // [CODE x CBSZ]
    convT(align_w,  align_fT, HID,  CODE);  // [CODE x HID]
    conv (align_w,  align_bT, (size_t)HID * CODE);   // bwd Wt = align_w itself
    convT(drift_w1, drift1T,  HID,  HID);
    convT(drift_w2, drift2T,  HID,  HID);
    convT(head_w,   head_wT,  HID,  NCLS);  // [NCLS x HID]
    convT(gate_w1,  gate_w1T, FEAT, GATEH); // [GATEH x FEAT]
    conv (proto,    protoB,   (size_t)P * HID);      // dots Wt = proto itself
    conv (x, xb, (size_t)Bsz * DIN);
    proto_norm_kernel<<<(P + 255) / 256, 256, 0, stream>>>(proto, pn, P, HID);

    // ---- encoder
    gemm(xb, DIN, enc_w1T, enc_b1, nullptr, 0, nullptr, 0, t0r, HID,
         HID, DIN, 1.0f, 1, 0);                                   // t0 = relu
    gemm(t0r, HID, enc_w2T, enc_b2, nullptr, 0, tcd, HID, nullptr, 0,
         HID, HID, 1.0f, 1, 0);                                   // t1 = relu
    layernorm_kernel<<<Bsz, 256, 0, stream>>>(tcd, enc_ln_g, enc_ln_b, HID,
        h_f, hb, HID, featb, FEAT);                               // h0 (f32,bf16,feat)

    // ---- symbolizer
    gemm(xb, DIN, sym_wT, sym_b, nullptr, 0, s_f, CODE, nullptr, 0,
         CODE, DIN, 1.0f, 1, 0);                                  // hs = relu
    layernorm_kernel<<<Bsz, 256, 0, stream>>>(s_f, sym_ln_g, sym_ln_b, CODE,
        nullptr, hp_bf, CODE, nullptr, 0);                        // hp (bf16)
    gemm(hp_bf, CODE, cb_wT, cb_b, nullptr, 0, tcd, CBSZ, nullptr, 0,
         CBSZ, CODE, 1.0f, 0, 0);                                 // codebook logits
    softmax_kernel<<<Bsz, 256, 0, stream>>>(tcd, tau, piu);       // pi (bf16)
    gemm(piu, CBSZ, cbookT, nullptr, nullptr, 0, s_f, CODE, nullptr, 0,
         CODE, CBSZ, 1.0f, 0, 0);                                 // s = pi @ codebook

    // ---- investment gate
    gemm(hb, HID, head_wT, head_b, nullptr, 0, nullptr, 0,
         featb + HID, FEAT, NCLS, HID, 1.0f, 0, 0);               // logits_mlp -> feat
    gemm(featb, FEAT, gate_w1T, gate_b1, nullptr, 0, gh_f, GATEH,
         nullptr, 0, GATEH, FEAT, 1.0f, 1, 0);                    // gate hidden
    effort_kernel<<<(Bsz + 255) / 256, 256, 0, stream>>>(gh_f, gate_w2, gate_b2,
        eff, Bsz, GATEH);

    // ---- energy-minimization loop
    const float decays[3] = {1.0f, 0.36787944117144233f, 0.1353352832366127f};
    const float calign = 2.0f * 0.5f / (float)CODE;               // 2*LAM_ALIGN/CODE
    for (int t = 0; t < 3; ++t) {
        gemm(hb, HID, align_fT, align_b, s_f, CODE, nullptr, 0,
             t0r, CODE, CODE, HID, 1.0f, 0, 0);                   // r = a - s (bf16)
        gemm(t0r, CODE, align_bT, nullptr, nullptr, 0, gbuf, HID,
             nullptr, 0, HID, CODE, calign, 0, 0);                // gbuf = calign*r@Wt
        gemm(hb, HID, protoB, nullptr, nullptr, 0, tcd, P, nullptr, 0,
             P, HID, 1.0f, 0, 0);                                 // dots = h@protoT
        argmin_kernel<<<Bsz, 256, 0, stream>>>(tcd, pn, jmin, P);
        gemm(hb, HID, drift1T, drift_b1, nullptr, 0, nullptr, 0,
             piu, HID, HID, HID, 1.0f, 2, 0);                     // u = tanh (bf16)
        gemm(piu, HID, drift2T, drift_b2, nullptr, 0, gbuf, HID,
             nullptr, 0, HID, HID, 1.0f, 0, 1);                   // gbuf += drift
        update_kernel<<<(int)(((size_t)Bsz * HID) / 256), 256, 0, stream>>>(
            h_f, hb, gbuf, jmin, proto, eff, steps, t, decays[t]);
    }

    // ---- outputs
    gemm(hb, HID, head_wT, head_b, nullptr, 0, lr_f, NCLS, nullptr, 0,
         NCLS, HID, 1.0f, 0, 0);                                  // logits_refined
    gemm(hb, HID, protoB, nullptr, nullptr, 0, tcd, P, nullptr, 0,
         P, HID, 1.0f, 0, 0);                                     // dots(final)
    final_kernel<<<Bsz, 128, 0, stream>>>(h_f, tcd, pn, lr_f, out);
}
